// PairBiasedSelfAttention_66520453481111
// MI455X (gfx1250) — compile-verified, matrix operand order checked by reading, not run on hardware
//
#include <hip/hip_runtime.h>

// ---------------- problem constants ----------------
#define BB 2
#define LL 1024
#define CS 384
#define CZ 128
#define HH 12
#define DD 32
#define NEG (-10000.0f)
#define SCALE 0.17677669529663687f  // 32^-0.5
#define NC 4                        // attention j-chunks (flash split)

typedef __attribute__((ext_vector_type(16))) __bf16          v16bf;
typedef __attribute__((ext_vector_type(8)))  float           v8f;
typedef __attribute__((ext_vector_type(16))) unsigned short  v16u;
typedef __attribute__((ext_vector_type(4)))  float           fvec4;

union ABTile {
    uint4 q[2];   // two 16B loads = 16 bf16 halves, slot order matches v16bf
    v16u  u;      // per-element packing
    v16bf b;      // WMMA operand
};

static __device__ __forceinline__ unsigned short f2bf(float f) {
    __bf16 h = (__bf16)f;                       // hardware v_cvt_*bf16_f32 (RNE)
    return __builtin_bit_cast(unsigned short, h);
}
static __device__ __forceinline__ float bf2f(unsigned short u) {
    return (float)__builtin_bit_cast(__bf16, u);
}
static __device__ __forceinline__ v8f wmma_bf16(v16bf a, v16bf b, v8f c) {
    return __builtin_amdgcn_wmma_f32_16x16x32_bf16(false, a, false, b, (short)0, c, false, false);
}

// ---------------- kernel 1: layernorm -> x bf16 ----------------
__global__ void k_layernorm(const float* __restrict__ s,
                            const float* __restrict__ gamma,
                            const float* __restrict__ beta,
                            unsigned short* __restrict__ xb) {
    __shared__ float r1[128], r2[128];
    const int row = blockIdx.x;            // B*L rows
    const int t = threadIdx.x;             // 128 threads, 3 elems each
    const float* sr = s + (size_t)row * CS;
    float a0 = sr[t], a1 = sr[t + 128], a2 = sr[t + 256];
    r1[t] = a0 + a1 + a2;
    r2[t] = a0 * a0 + a1 * a1 + a2 * a2;
    __syncthreads();
    for (int off = 64; off > 0; off >>= 1) {
        if (t < off) { r1[t] += r1[t + off]; r2[t] += r2[t + off]; }
        __syncthreads();
    }
    const float mu = r1[0] * (1.0f / CS);
    const float var = r2[0] * (1.0f / CS) - mu * mu;
    const float rs = rsqrtf(var + 1e-5f);
    unsigned short* xr = xb + (size_t)row * CS;
    xr[t]       = f2bf((a0 - mu) * rs * gamma[t]       + beta[t]);
    xr[t + 128] = f2bf((a1 - mu) * rs * gamma[t + 128] + beta[t + 128]);
    xr[t + 256] = f2bf((a2 - mu) * rs * gamma[t + 256] + beta[t + 256]);
}

// ---------------- kernel 2: weights -> bf16 (Wp padded to 16 heads) ----------------
__global__ void k_wcvt(const float* __restrict__ Wq, const float* __restrict__ Wk,
                       const float* __restrict__ Wv, const float* __restrict__ Wo,
                       const float* __restrict__ Wp,
                       unsigned short* __restrict__ wqb, unsigned short* __restrict__ wkb,
                       unsigned short* __restrict__ wvb, unsigned short* __restrict__ wob,
                       unsigned short* __restrict__ wpb) {
    const int N = 4 * CS * CS + 16 * CZ;
    int idx = blockIdx.x * blockDim.x + threadIdx.x;
    if (idx >= N) return;
    const int W2 = CS * CS;
    if (idx < W2)            { wqb[idx] = f2bf(Wq[idx]); return; }
    if (idx < 2 * W2)        { wkb[idx - W2] = f2bf(Wk[idx - W2]); return; }
    if (idx < 3 * W2)        { wvb[idx - 2 * W2] = f2bf(Wv[idx - 2 * W2]); return; }
    if (idx < 4 * W2)        { wob[idx - 3 * W2] = f2bf(Wo[idx - 3 * W2]); return; }
    int rem = idx - 4 * W2;  // 16x128 padded Wp
    int h = rem / CZ, c = rem % CZ;
    wpb[rem] = (h < HH) ? f2bf(Wp[h * CZ + c]) : 0;
}

// ---------------- kernel 3: q/k/v projections (WMMA bf16); v stored transposed ----------------
__global__ void k_qkv(const unsigned short* __restrict__ xb,
                      const unsigned short* __restrict__ wqb,
                      const unsigned short* __restrict__ wkb,
                      const unsigned short* __restrict__ wvb,
                      const float* __restrict__ bq, const float* __restrict__ bk,
                      const float* __restrict__ bv,
                      unsigned short* __restrict__ qw, unsigned short* __restrict__ kw,
                      unsigned short* __restrict__ vtw) {
    const int MT = (BB * LL) / 16, NT = CS / 16;           // 128, 24
    const int wid = blockIdx.x * (blockDim.x >> 5) + (threadIdx.x >> 5);
    if (wid >= 3 * MT * NT) return;
    const int p = wid / (MT * NT);
    const int rem = wid % (MT * NT);
    const int mt = rem / NT, nt = rem % NT;
    const int l = threadIdx.x & 31, g = l >> 4, ln = l & 15;

    const unsigned short* wsel = (p == 0) ? wqb : (p == 1) ? wkb : wvb;
    const float* bsel = (p == 0) ? bq : (p == 1) ? bk : bv;

    const int m = mt * 16 + ln;
    const int n = nt * 16 + ln;
    const uint4* arow = (const uint4*)(xb + (size_t)m * CS);      // 48 uint4/row
    const uint4* brow = (const uint4*)(wsel + (size_t)n * CS);

    v8f c = {0, 0, 0, 0, 0, 0, 0, 0};
#pragma unroll
    for (int kc = 0; kc < CS / 32; ++kc) {
        ABTile a, b;
        a.q[0] = arow[kc * 4 + g]; a.q[1] = arow[kc * 4 + g + 2];
        b.q[0] = brow[kc * 4 + g]; b.q[1] = brow[kc * 4 + g + 2];
        c = wmma_bf16(a.b, b.b, c);
    }
    const float bias = bsel[n];
    const int h = n >> 5, d = n & 31;
#pragma unroll
    for (int r = 0; r < 8; ++r) {
        const int gm = mt * 16 + r + 8 * g;                  // global row in B*L
        const int bidx = gm >> 10, lrow = gm & (LL - 1);
        const unsigned short val = f2bf(c[r] + bias);
        if (p == 0) {
            qw[(((size_t)bidx * HH + h) * LL + lrow) * DD + d] = val;
        } else if (p == 1) {
            kw[(((size_t)bidx * HH + h) * LL + lrow) * DD + d] = val;
        } else {
            // v transposed: [b][h][d][l] so PV B-operand loads are contiguous
            vtw[(((size_t)bidx * HH + h) * DD + d) * LL + lrow] = val;
        }
    }
}

// ---------------- kernel 4: pair bias GEMM: z(fp32->bf16, NT) x Wp^T, WMMA ----------------
__global__ void k_bias(const float* __restrict__ z,
                       const unsigned short* __restrict__ wpb,
                       unsigned short* __restrict__ biasb) {
    const int JT = LL / 16;                                  // 64
    const int wid = blockIdx.x * (blockDim.x >> 5) + (threadIdx.x >> 5);
    if (wid >= BB * LL * JT) return;
    const int jt = wid % JT;
    const int i = (wid / JT) % LL;
    const int b = wid / (JT * LL);
    const int l = threadIdx.x & 31, g = l >> 4, ln = l & 15;

    const int j = jt * 16 + ln;                              // A row = pair (i,j)
    const float* zrow = z + (((size_t)b * LL + i) * LL + j) * CZ;
    const uint4* brow = (const uint4*)(wpb + (size_t)ln * CZ);  // lane = head col

    v8f c = {0, 0, 0, 0, 0, 0, 0, 0};
#pragma unroll
    for (int kc = 0; kc < CZ / 32; ++kc) {
        const int k0 = kc * 32 + g * 8;
        fvec4 f0 = __builtin_nontemporal_load((const fvec4*)(zrow + k0));
        fvec4 f1 = __builtin_nontemporal_load((const fvec4*)(zrow + k0 + 4));
        fvec4 f2 = __builtin_nontemporal_load((const fvec4*)(zrow + k0 + 16));
        fvec4 f3 = __builtin_nontemporal_load((const fvec4*)(zrow + k0 + 20));
        ABTile a, b;
        a.u[0] = f2bf(f0.x); a.u[1] = f2bf(f0.y); a.u[2] = f2bf(f0.z); a.u[3] = f2bf(f0.w);
        a.u[4] = f2bf(f1.x); a.u[5] = f2bf(f1.y); a.u[6] = f2bf(f1.z); a.u[7] = f2bf(f1.w);
        a.u[8] = f2bf(f2.x); a.u[9] = f2bf(f2.y); a.u[10] = f2bf(f2.z); a.u[11] = f2bf(f2.w);
        a.u[12] = f2bf(f3.x); a.u[13] = f2bf(f3.y); a.u[14] = f2bf(f3.z); a.u[15] = f2bf(f3.w);
        b.q[0] = brow[kc * 4 + g]; b.q[1] = brow[kc * 4 + g + 2];
        c = wmma_bf16(a.b, b.b, c);
    }
    const int h = ln;
    if (h < HH) {
        // lane holds 8 consecutive j values (rows r+8g of C) for one head
        uint4 st;
        st.x = (unsigned)f2bf(c[0]) | ((unsigned)f2bf(c[1]) << 16);
        st.y = (unsigned)f2bf(c[2]) | ((unsigned)f2bf(c[3]) << 16);
        st.z = (unsigned)f2bf(c[4]) | ((unsigned)f2bf(c[5]) << 16);
        st.w = (unsigned)f2bf(c[6]) | ((unsigned)f2bf(c[7]) << 16);
        *(uint4*)(biasb + (((size_t)(b * HH + h) * LL + i) * LL + jt * 16 + g * 8)) = st;
    }
}

// ---------------- kernel 5: flash attention (per j-chunk), writes partials ----------------
__global__ void k_attn(const unsigned short* __restrict__ qw,
                       const unsigned short* __restrict__ kw,
                       const unsigned short* __restrict__ vtw,
                       const unsigned short* __restrict__ biasb,
                       const unsigned char* __restrict__ maskb,
                       float* __restrict__ pO, float* __restrict__ pm,
                       float* __restrict__ pl_) {
    __shared__ unsigned short Plds[8][16 * 32];              // per-wave P staging
    const int IT = LL / 16;                                  // 64
    const int wid = blockIdx.x * (blockDim.x >> 5) + (threadIdx.x >> 5);
    if (wid >= BB * HH * IT * NC) return;
    const int ch = wid % NC;
    const int t0 = wid / NC;
    const int it = t0 % IT;
    const int h = (t0 / IT) % HH;
    const int b = t0 / (IT * HH);
    const int l = threadIdx.x & 31, g = l >> 4, ln = l & 15;
    const int wslot = (threadIdx.x >> 5);

    const size_t headbase = ((size_t)b * HH + h) * LL;
    // A = q tile (16 x 32), lane row = i-local
    const uint4* qrow = (const uint4*)(qw + (headbase + it * 16 + ln) * DD);
    ABTile aq; aq.q[0] = qrow[g]; aq.q[1] = qrow[g + 2];

    float mrow[8], lrow[8];
#pragma unroll
    for (int r = 0; r < 8; ++r) { mrow[r] = -3.0e38f; lrow[r] = 0.0f; }
    v8f acc0 = {0, 0, 0, 0, 0, 0, 0, 0}, acc1 = {0, 0, 0, 0, 0, 0, 0, 0};

    const unsigned short* biasrow = biasb + (headbase + it * 16) * LL;
    unsigned short* pl = &Plds[wslot][0];

    for (int tt = 0; tt < (LL / 16) / NC; ++tt) {            // 16 j-tiles per chunk
        const int jt = ch * ((LL / 16) / NC) + tt;
        const int j0 = jt * 16, jn = j0 + ln;
        // B = k^T tile (32 x 16): lane col = j-local, slots = d
        const uint4* krow = (const uint4*)(kw + (headbase + jn) * DD);
        ABTile bk_; bk_.q[0] = krow[g]; bk_.q[1] = krow[g + 2];
        v8f zero = {0, 0, 0, 0, 0, 0, 0, 0};
        v8f sc = wmma_bf16(aq.b, bk_.b, zero);

        const float km = maskb[b * LL + jn] ? 0.0f : NEG;    // key mask (column)
        float lg[8], p[8];
#pragma unroll
        for (int r = 0; r < 8; ++r) {
            const unsigned short bv =
                __builtin_nontemporal_load(biasrow + (size_t)(r + 8 * g) * LL + jn);
            lg[r] = sc[r] * SCALE + bf2f(bv) + km;
        }
        // per-row tile max across the 16-lane half (rows r+8g)
        float mnew[8], alpha[8];
#pragma unroll
        for (int r = 0; r < 8; ++r) {
            float t = lg[r];
            t = fmaxf(t, __shfl_xor(t, 1, 32));
            t = fmaxf(t, __shfl_xor(t, 2, 32));
            t = fmaxf(t, __shfl_xor(t, 4, 32));
            t = fmaxf(t, __shfl_xor(t, 8, 32));
            mnew[r] = fmaxf(mrow[r], t);
            alpha[r] = __expf(mrow[r] - mnew[r]);
            p[r] = __expf(lg[r] - mnew[r]);
            float rs = p[r];
            rs += __shfl_xor(rs, 1, 32);
            rs += __shfl_xor(rs, 2, 32);
            rs += __shfl_xor(rs, 4, 32);
            rs += __shfl_xor(rs, 8, 32);
            lrow[r] = lrow[r] * alpha[r] + rs;
            mrow[r] = mnew[r];
            acc0[r] *= alpha[r];
            acc1[r] *= alpha[r];
        }
        // stage P (C layout -> memory) for A-layout reload
#pragma unroll
        for (int r = 0; r < 8; ++r)
            pl[(r + 8 * g) * 32 + (tt & 1) * 16 + ln] = f2bf(p[r]);
        asm volatile("" ::: "memory");                       // keep store->load order

        if (tt & 1) {
            // A = P (16 x 32): lane row = i-local, slots = j-local
            const uint4* pr = (const uint4*)&pl[ln * 32];
            ABTile ap; ap.q[0] = pr[g]; ap.q[1] = pr[g + 2];
            // B = v^T (32 x 16): lane col = d, slots = j (contiguous in vtw)
            const int jp = (jt - 1) * 16;                    // 32-wide j window start
            const size_t vtbase = ((size_t)b * HH + h) * DD * LL;
            const uint4* v0 = (const uint4*)(vtw + vtbase + (size_t)ln * LL + jp);
            const uint4* v1 = (const uint4*)(vtw + vtbase + (size_t)(16 + ln) * LL + jp);
            ABTile bv0, bv1;
            bv0.q[0] = v0[g]; bv0.q[1] = v0[g + 2];
            bv1.q[0] = v1[g]; bv1.q[1] = v1[g + 2];
            acc0 = wmma_bf16(ap.b, bv0.b, acc0);
            acc1 = wmma_bf16(ap.b, bv1.b, acc1);
        }
    }
    // write partials (fp32): O in C layout, per-row m/l once per row
#pragma unroll
    for (int r = 0; r < 8; ++r) {
        const int row = r + 8 * g;
        pO[((size_t)wid * 16 + row) * 32 + ln] = acc0[r];
        pO[((size_t)wid * 16 + row) * 32 + 16 + ln] = acc1[r];
    }
    if (ln < 8) {
        pm[(size_t)wid * 16 + 8 * g + ln] = mrow[ln];
        pl_[(size_t)wid * 16 + 8 * g + ln] = lrow[ln];
    }
}

// ---------------- kernel 6: combine flash partials -> o (bf16, [b][l][h*32+d]) ----------
__global__ void k_combine(const float* __restrict__ pO, const float* __restrict__ pm,
                          const float* __restrict__ pl_, unsigned short* __restrict__ ob) {
    const int IT = LL / 16;
    const int idx = blockIdx.x * blockDim.x + threadIdx.x;   // one output element
    if (idx >= BB * HH * IT * 16 * 32) return;
    const int col = idx & 31;
    const int row = (idx >> 5) & 15;
    const int t0 = idx >> 9;                                 // (b,h,it)
    const int it = t0 % IT;
    const int h = (t0 / IT) % HH;
    const int b = t0 / (IT * HH);

    float mc[NC];
    float M = -3.0e38f;
#pragma unroll
    for (int c = 0; c < NC; ++c) {
        mc[c] = pm[(size_t)(t0 * NC + c) * 16 + row];
        M = fmaxf(M, mc[c]);
    }
    float lsum = 0.0f, osum = 0.0f;
#pragma unroll
    for (int c = 0; c < NC; ++c) {
        const float w = __expf(mc[c] - M);
        lsum += pl_[(size_t)(t0 * NC + c) * 16 + row] * w;
        osum += pO[((size_t)(t0 * NC + c) * 16 + row) * 32 + col] * w;
    }
    const int grow = it * 16 + row;
    ob[((size_t)b * LL + grow) * CS + h * DD + col] = f2bf(osum / lsum);
}

// ---------------- kernel 7: output projection + residue mask (fp32 out, NT) ----------
__global__ void k_oproj(const unsigned short* __restrict__ ob,
                        const unsigned short* __restrict__ wob,
                        const float* __restrict__ bo,
                        const unsigned char* __restrict__ maskb,
                        float* __restrict__ out) {
    const int MT = (BB * LL) / 16, NT = CS / 16;
    const int wid = blockIdx.x * (blockDim.x >> 5) + (threadIdx.x >> 5);
    if (wid >= MT * NT) return;
    const int mt = wid / NT, nt = wid % NT;
    const int l = threadIdx.x & 31, g = l >> 4, ln = l & 15;

    const int m = mt * 16 + ln, n = nt * 16 + ln;
    const uint4* arow = (const uint4*)(ob + (size_t)m * CS);
    const uint4* brow = (const uint4*)(wob + (size_t)n * CS);

    v8f c = {0, 0, 0, 0, 0, 0, 0, 0};
#pragma unroll
    for (int kc = 0; kc < CS / 32; ++kc) {
        ABTile a, b;
        a.q[0] = arow[kc * 4 + g]; a.q[1] = arow[kc * 4 + g + 2];
        b.q[0] = brow[kc * 4 + g]; b.q[1] = brow[kc * 4 + g + 2];
        c = wmma_bf16(a.b, b.b, c);
    }
    const float bias = bo[n];
#pragma unroll
    for (int r = 0; r < 8; ++r) {
        const int gm = mt * 16 + r + 8 * g;
        const float mk = maskb[gm] ? 1.0f : 0.0f;
        __builtin_nontemporal_store((c[r] + bias) * mk, out + (size_t)gm * CS + n);
    }
}

// ---------------- host launcher ----------------
extern "C" void kernel_launch(void* const* d_in, const int* in_sizes, int n_in,
                              void* d_out, int out_size, void* d_ws, size_t ws_size,
                              hipStream_t stream) {
    const float* s     = (const float*)d_in[0];
    const float* z     = (const float*)d_in[1];
    const unsigned char* mask = (const unsigned char*)d_in[2]; // jnp.bool_
    const float* Wq = (const float*)d_in[3];  const float* bq = (const float*)d_in[4];
    const float* Wk = (const float*)d_in[5];  const float* bk = (const float*)d_in[6];
    const float* Wv = (const float*)d_in[7];  const float* bv = (const float*)d_in[8];
    const float* Wo = (const float*)d_in[9];  const float* bo = (const float*)d_in[10];
    const float* Wp = (const float*)d_in[11];
    const float* gamma = (const float*)d_in[12];
    const float* beta  = (const float*)d_in[13];
    float* out = (float*)d_out;

    // workspace carve-up
    char* base = (char*)d_ws;
    size_t off = 0;
    auto alloc = [&](size_t bytes) { char* p = base + off; off += (bytes + 255) & ~(size_t)255; return p; };
    unsigned short* xb  = (unsigned short*)alloc((size_t)BB * LL * CS * 2);
    unsigned short* wqb = (unsigned short*)alloc((size_t)CS * CS * 2);
    unsigned short* wkb = (unsigned short*)alloc((size_t)CS * CS * 2);
    unsigned short* wvb = (unsigned short*)alloc((size_t)CS * CS * 2);
    unsigned short* wob = (unsigned short*)alloc((size_t)CS * CS * 2);
    unsigned short* wpb = (unsigned short*)alloc((size_t)16 * CZ * 2);
    unsigned short* qw  = (unsigned short*)alloc((size_t)BB * HH * LL * DD * 2);
    unsigned short* kw  = (unsigned short*)alloc((size_t)BB * HH * LL * DD * 2);
    unsigned short* vtw = (unsigned short*)alloc((size_t)BB * HH * LL * DD * 2);
    unsigned short* ob  = (unsigned short*)alloc((size_t)BB * LL * CS * 2);
    const size_t ntask = (size_t)BB * HH * (LL / 16) * NC;   // 6144 flash partial tasks
    float* pO  = (float*)alloc(ntask * 16 * 32 * sizeof(float));   // ~12.6 MB
    float* pm  = (float*)alloc(ntask * 16 * sizeof(float));
    float* pl_ = (float*)alloc(ntask * 16 * sizeof(float));
    unsigned short* biasb = (unsigned short*)alloc((size_t)BB * HH * LL * LL * 2); // 50 MB

    // 1) layernorm
    k_layernorm<<<BB * LL, 128, 0, stream>>>(s, gamma, beta, xb);
    // 2) weight conversion
    {
        int N = 4 * CS * CS + 16 * CZ;
        k_wcvt<<<(N + 255) / 256, 256, 0, stream>>>(Wq, Wk, Wv, Wo, Wp, wqb, wkb, wvb, wob, wpb);
    }
    // 3) q/k/v projections: 3 * 128 * 24 waves
    k_qkv<<<(3 * 128 * 24) / 8, 256, 0, stream>>>(xb, wqb, wkb, wvb, bq, bk, bv, qw, kw, vtw);
    // 4) pair bias: B*L*(L/16) waves = 131072 (z streamed once, NT)
    k_bias<<<(BB * LL * (LL / 16)) / 8, 256, 0, stream>>>(z, wpb, biasb);
    // 5) flash attention partials: B*H*(L/16)*NC waves = 6144
    k_attn<<<(BB * HH * (LL / 16) * NC) / 8, 256, 0, stream>>>(qw, kw, vtw, biasb, mask,
                                                               pO, pm, pl_);
    // 6) combine partials: one thread per output element
    {
        int N = BB * HH * (LL / 16) * 16 * 32;
        k_combine<<<(N + 255) / 256, 256, 0, stream>>>(pO, pm, pl_, ob);
    }
    // 7) output projection: 128*24 waves
    k_oproj<<<(128 * 24) / 8, 256, 0, stream>>>(ob, wob, bo, mask, out);
}
